// SAConvLSTMCell_20495583937284
// MI455X (gfx1250) — matrix operand order repaired and checked by reading, not compiled, not run on hardware
//
#include <hip/hip_runtime.h>

// ---------------------------------------------------------------------------
// SA-ConvLSTM cell for gfx1250 (MI455X).
// B=16, Cin=Ch=64, A=32, H=W=32, HW=1024, M=B*HW=16384, Kconv=128*9=1152.
// Matmuls (2x conv-as-GEMM, QK^T, PV) run on v_wmma_f32_16x16x32_bf16 with
// f32 accumulation; attention is fused (scores live in 64KB LDS per tile).
// ---------------------------------------------------------------------------

typedef __attribute__((ext_vector_type(16))) __bf16 v16bf;
typedef __attribute__((ext_vector_type(8)))  __bf16 v8bf;
typedef __attribute__((ext_vector_type(8)))  float  v8f;

#define WMMA_BF16(a, b, c) \
  __builtin_amdgcn_wmma_f32_16x16x32_bf16(false, (a), false, (b), (short)0, (c), false, false)

__device__ __forceinline__ float sigm(float x) { return 1.f / (1.f + __expf(-x)); }

// A-fragment (16x32 bf16): lane row = lane&15; element e -> K=e+16*hi.
// Two contiguous 8-half (16B) runs per lane.
__device__ __forceinline__ v16bf load_a_frag(const __bf16* row, int hi) {
  v8bf lo = *(const v8bf*)(row + 16 * hi);
  v8bf h8 = *(const v8bf*)(row + 16 * hi + 8);
  v16bf a;
#pragma unroll
  for (int i = 0; i < 8; ++i) { a[i] = lo[i]; a[8 + i] = h8[i]; }
  return a;
}

// ---------------------------------------------------------------------------
// f32 -> bf16 flat convert (weight packing; conv_w layout [O][Ci][3][3] is
// already K-contiguous for our im2col K ordering k = ci*9+ky*3+kx).
// ---------------------------------------------------------------------------
__global__ void k_cvt_bf16(const float* __restrict__ src, __bf16* __restrict__ dst, int n) {
  int t = blockIdx.x * 256 + threadIdx.x;
  if (t < n) dst[t] = (__bf16)src[t];
}

// ---------------------------------------------------------------------------
// im2col: out[m][k] bf16, m=(b,y,x), k=(ci,ky,kx); ci<64 from src0 else src1.
// ---------------------------------------------------------------------------
__global__ void k_im2col(const float* __restrict__ src0, const float* __restrict__ src1,
                         __bf16* __restrict__ out) {
  long long t = (long long)blockIdx.x * 256 + threadIdx.x;  // 16384*1152
  if (t >= (long long)16384 * 1152) return;
  int k = (int)(t % 1152);
  int mrow = (int)(t / 1152);
  int kx = k % 3, ky = (k / 3) % 3, ci = k / 9;
  int x = mrow & 31, y = (mrow >> 5) & 31, b = mrow >> 10;
  int yy = y + ky - 1, xx = x + kx - 1;
  float v = 0.f;
  if (yy >= 0 && yy < 32 && xx >= 0 && xx < 32) {
    const float* s = (ci < 64) ? src0 : src1;
    int cc = (ci < 64) ? ci : ci - 64;
    v = s[(((size_t)b * 64 + cc) << 10) + (yy << 5) + xx];
  }
  out[t] = (__bf16)v;
}

// ---------------------------------------------------------------------------
// bf16 GEMM: C[M x 256] = A[M x K] * Bt^T, Bt stored [N][K] (K-contiguous).
// Block = 128 threads (4 waves); block covers 32 rows x 256 cols;
// each wave computes 32x64 via 8 accumulators (2 A-frags x 4 B-frags per
// k-step -> 8 WMMAs per 6 fragment loads). grid.x = M/32.
// A rows are a 37.7MB single-pass stream -> prefetch ahead into GL2.
// ---------------------------------------------------------------------------
__global__ __launch_bounds__(128) void k_gemm_bf16(const __bf16* __restrict__ A,
                                                   const __bf16* __restrict__ Bt,
                                                   float* __restrict__ C,
                                                   int Kdim, int lda, int ldb, int ldc) {
  int lane = threadIdx.x & 31, wave = threadIdx.x >> 5;
  int hi = lane >> 4, l15 = lane & 15;
  int m0 = blockIdx.x * 32;
  const __bf16* arow0 = A + (size_t)(m0 + l15) * lda;
  const __bf16* arow1 = A + (size_t)(m0 + 16 + l15) * lda;
  v8f acc0[4] = {};
  v8f acc1[4] = {};
  for (int k0 = 0; k0 < Kdim; k0 += 32) {
    __builtin_prefetch(arow0 + k0 + 64, 0, 3);  // global_prefetch_b8, 2 steps ahead
    __builtin_prefetch(arow1 + k0 + 64, 0, 3);
    v16bf a0 = load_a_frag(arow0 + k0, hi);
    v16bf a1 = load_a_frag(arow1 + k0, hi);
#pragma unroll
    for (int t = 0; t < 4; ++t) {
      int n = wave * 64 + t * 16 + l15;
      v16bf b = *(const v16bf*)(Bt + (size_t)n * ldb + k0 + 16 * hi);
      acc0[t] = WMMA_BF16(a0, b, acc0[t]);
      acc1[t] = WMMA_BF16(a1, b, acc1[t]);
    }
  }
#pragma unroll
  for (int t = 0; t < 4; ++t) {
    int n = wave * 64 + t * 16 + l15;
#pragma unroll
    for (int r = 0; r < 8; ++r) {
      C[(size_t)(m0 + r + 8 * hi) * ldc + n]      = acc0[t][r];
      C[(size_t)(m0 + 16 + r + 8 * hi) * ldc + n] = acc1[t][r];
    }
  }
}

// ---------------------------------------------------------------------------
// Per-(b, cout) GroupNorm stats over HW=1024. One wave per channel.
// Conv bias cancels exactly under per-channel GN, so GEMM output is bias-free.
// ---------------------------------------------------------------------------
__global__ __launch_bounds__(128) void k_gnstats(const float* __restrict__ C,
                                                 float* __restrict__ stats) {
  int lane = threadIdx.x & 31, wave = threadIdx.x >> 5;
  int gid = blockIdx.x * 4 + wave;  // 4096 = 16 * 256
  int b = gid >> 8, o = gid & 255;
  const float* base = C + ((size_t)b << 10) * 256 + o;
  float s = 0.f, s2 = 0.f;
  for (int j = lane; j < 1024; j += 32) {
    float v = base[(size_t)j * 256];
    s += v; s2 += v * v;
  }
  for (int msk = 1; msk < 32; msk <<= 1) {
    s  += __shfl_xor(s,  msk, 32);
    s2 += __shfl_xor(s2, msk, 32);
  }
  if (lane == 0) {
    float mean = s * (1.f / 1024.f);
    float var  = s2 * (1.f / 1024.f) - mean * mean;
    stats[2 * gid]     = mean;
    stats[2 * gid + 1] = rsqrtf(var + 1e-5f);
  }
}

// ---------------------------------------------------------------------------
// Gate block 1: c_next = sig(f)*c + sig(i)*tanh(g); also keep sig(o).
// ---------------------------------------------------------------------------
__global__ void k_gate1(const float* __restrict__ conv, const float* __restrict__ stats,
                        const float* __restrict__ gw, const float* __restrict__ gb,
                        const float* __restrict__ c_in,
                        float* __restrict__ c1, float* __restrict__ sco) {
  size_t t = (size_t)blockIdx.x * 256 + threadIdx.x;  // 1048576
  int p = (int)(t & 1023), ch = (int)((t >> 10) & 63), b = (int)(t >> 16);
  size_t mr = ((size_t)(b << 10) + p) * 256;
  const float* st = stats + (size_t)b * 512;
  float g[4];
#pragma unroll
  for (int q = 0; q < 4; ++q) {
    int o = ch + q * 64;
    g[q] = (conv[mr + o] - st[2 * o]) * st[2 * o + 1] * gw[o] + gb[o];
  }
  float cn = sigm(g[1]) * c_in[t] + sigm(g[0]) * tanhf(g[3]);
  c1[t] = cn;
  sco[t] = sigm(g[2]);
}

// ---------------------------------------------------------------------------
// Gate block 2: lc_next, c_next (outputs), h_mid.
// ---------------------------------------------------------------------------
__global__ void k_gate2(const float* __restrict__ conv, const float* __restrict__ stats,
                        const float* __restrict__ gw, const float* __restrict__ gb,
                        const float* __restrict__ lc_in, const float* __restrict__ sco,
                        float* __restrict__ out_c, float* __restrict__ out_lc,
                        float* __restrict__ hmid) {
  size_t t = (size_t)blockIdx.x * 256 + threadIdx.x;
  int p = (int)(t & 1023), ch = (int)((t >> 10) & 63), b = (int)(t >> 16);
  size_t mr = ((size_t)(b << 10) + p) * 256;
  const float* st = stats + (size_t)b * 512;
  float g[4];
#pragma unroll
  for (int q = 0; q < 4; ++q) {
    int o = ch + q * 64;
    g[q] = (conv[mr + o] - st[2 * o]) * st[2 * o + 1] * gw[o] + gb[o];
  }
  float lcn = sigm(g[1]) * lc_in[t] + sigm(g[0]) * tanhf(g[3]);
  float cn2 = sigm(g[2]) * tanhf(lcn);
  out_lc[t] = lcn;
  out_c[t]  = cn2;
  hmid[t]   = sco[t] * tanhf(cn2);
}

// ---------------------------------------------------------------------------
// Q/K projections: Qt/Kt/Kmt stored [b][p][32] bf16 (A-dim contiguous).
// ---------------------------------------------------------------------------
__global__ void k_qk(const float* __restrict__ hmid, const float* __restrict__ m_in,
                     const float* __restrict__ wq, const float* __restrict__ bq,
                     const float* __restrict__ wk, const float* __restrict__ bk,
                     const float* __restrict__ wk2, const float* __restrict__ bk2,
                     __bf16* __restrict__ Qt, __bf16* __restrict__ Kt,
                     __bf16* __restrict__ Kmt) {
  int t = blockIdx.x * 128 + threadIdx.x;  // 524288
  int a = t & 31, p = (t >> 5) & 1023, b = t >> 15;
  const float* hp = hmid + ((size_t)b << 16) + p;
  const float* mp = m_in + ((size_t)b << 16) + p;
  float q = bq[a], kh = bk[a], km = bk2[a];
  for (int cc = 0; cc < 64; ++cc) {
    float hv = hp[(size_t)cc << 10], mv = mp[(size_t)cc << 10];
    q  += wq[a * 64 + cc] * hv;
    kh += wk[a * 64 + cc] * hv;
    km += wk2[a * 64 + cc] * mv;
  }
  size_t o = ((size_t)(b << 10) + p) * 32 + a;
  Qt[o] = (__bf16)q; Kt[o] = (__bf16)kh; Kmt[o] = (__bf16)km;
}

// ---------------------------------------------------------------------------
// V projections: Vh/Vm stored [b][c][q] bf16 (q contiguous: B-frag friendly).
// ---------------------------------------------------------------------------
__global__ void k_v(const float* __restrict__ hmid, const float* __restrict__ m_in,
                    const float* __restrict__ wv, const float* __restrict__ bv,
                    const float* __restrict__ wv2, const float* __restrict__ bv2,
                    __bf16* __restrict__ Vh, __bf16* __restrict__ Vm) {
  size_t t = (size_t)blockIdx.x * 256 + threadIdx.x;  // 1048576
  int p = (int)(t & 1023), c = (int)((t >> 10) & 63), b = (int)(t >> 16);
  const float* hp = hmid + ((size_t)b << 16) + p;
  const float* mp = m_in + ((size_t)b << 16) + p;
  float vh = bv[c], vm = bv2[c];
  for (int cc = 0; cc < 64; ++cc) {
    vh += wv[c * 64 + cc] * hp[(size_t)cc << 10];
    vm += wv2[c * 64 + cc] * mp[(size_t)cc << 10];
  }
  Vh[t] = (__bf16)vh; Vm[t] = (__bf16)vm;
}

// ---------------------------------------------------------------------------
// Fused attention: per (batch, 16-row p-tile). S = Q*K^T (K-dim = A = 32 ->
// ONE wmma per 16x16 tile), softmax over 1024 in LDS, Z = P*V via wmma.
// Block = 128 threads (4 waves), LDS = 64KB score strip.
// ---------------------------------------------------------------------------
__global__ __launch_bounds__(128) void k_attn(const __bf16* __restrict__ Qt,
                                              const __bf16* __restrict__ Kt,
                                              const __bf16* __restrict__ Vt,
                                              float* __restrict__ Z) {
  __shared__ float sc[16 * 1024];  // 64KB
  int lane = threadIdx.x & 31, wave = threadIdx.x >> 5;
  int hi = lane >> 4, l15 = lane & 15;
  int b = blockIdx.x >> 6, p0 = (blockIdx.x & 63) << 4;

  // ---- S = Q x K^T ----
  const __bf16* qrow = Qt + ((size_t)(b << 10) + p0 + l15) * 32;
  v16bf qa = load_a_frag(qrow, hi);
  for (int j = wave; j < 64; j += 4) {
    int q0 = j << 4;
    v16bf kb = *(const v16bf*)(Kt + ((size_t)(b << 10) + q0 + l15) * 32 + 16 * hi);
    v8f d = {};
    d = WMMA_BF16(qa, kb, d);
#pragma unroll
    for (int r = 0; r < 8; ++r)
      sc[(r + 8 * hi) * 1024 + q0 + l15] = d[r];
  }
  __syncthreads();

  // ---- softmax along q (8 threads per row, wave32 xor-butterfly) ----
  {
    int row = threadIdx.x >> 3, s = threadIdx.x & 7;
    float* rp = sc + row * 1024;
    int c0 = s * 128;
    float mx = -3.4e38f;
    for (int c = c0; c < c0 + 128; ++c) mx = fmaxf(mx, rp[c]);
    for (int msk = 1; msk < 8; msk <<= 1) mx = fmaxf(mx, __shfl_xor(mx, msk, 32));
    float sum = 0.f;
    for (int c = c0; c < c0 + 128; ++c) {
      float e = __expf(rp[c] - mx);
      rp[c] = e; sum += e;
    }
    for (int msk = 1; msk < 8; msk <<= 1) sum += __shfl_xor(sum, msk, 32);
    float inv = 1.f / sum;
    for (int c = c0; c < c0 + 128; ++c) rp[c] *= inv;
  }
  __syncthreads();

  // ---- Z = P x V^T  (wave w -> output cols w*16..+15) ----
  int c0 = wave << 4;
  const float* prow = sc + l15 * 1024;
  const __bf16* vrow = Vt + ((size_t)(b << 6) + c0 + l15) * 1024 + 16 * hi;
  v8f acc = {};
  for (int kq = 0; kq < 1024; kq += 32) {
    v16bf pa;
#pragma unroll
    for (int i = 0; i < 8; ++i) {
      pa[i]     = (__bf16)prow[kq + 16 * hi + i];
      pa[8 + i] = (__bf16)prow[kq + 16 * hi + 8 + i];
    }
    v16bf vb = *(const v16bf*)(vrow + kq);
    acc = WMMA_BF16(pa, vb, acc);
  }
#pragma unroll
  for (int r = 0; r < 8; ++r)
    Z[((size_t)(b << 10) + p0 + r + 8 * hi) * 64 + c0 + l15] = acc[r];
}

// ---------------------------------------------------------------------------
// Final 1x1 mixing + memory gates. One block per (b,p), 192 threads.
// wz (128x128), wm (192x192) stream from L2 (they fit easily in 192MB L2).
// ---------------------------------------------------------------------------
__global__ __launch_bounds__(192) void k_final(const float* __restrict__ Zh,
                                               const float* __restrict__ Zm,
                                               const float* __restrict__ hmid,
                                               const float* __restrict__ m_in,
                                               const float* __restrict__ wz,
                                               const float* __restrict__ bz,
                                               const float* __restrict__ wm,
                                               const float* __restrict__ bm,
                                               float* __restrict__ h_next,
                                               float* __restrict__ m_next) {
  __shared__ float zcat[128];
  __shared__ float zin[192];
  __shared__ float combs[192];
  int t = threadIdx.x;
  int b = blockIdx.x >> 10, p = blockIdx.x & 1023;
  size_t rowZ = (size_t)blockIdx.x * 64;
  if (t < 64)        zcat[t] = Zh[rowZ + t];
  else if (t < 128)  zcat[t] = Zm[rowZ + (t - 64)];
  else               zin[t]  = hmid[((size_t)b << 16) + ((size_t)(t - 128) << 10) + p];
  __syncthreads();
  if (t < 128) {
    float acc = bz[t];
    const float* wr = wz + t * 128;
    for (int j = 0; j < 128; ++j) acc += wr[j] * zcat[j];
    zin[t] = acc;
  }
  __syncthreads();
  {
    float acc = bm[t];
    const float* wr = wm + t * 192;
    for (int j = 0; j < 192; ++j) acc += wr[j] * zin[j];
    combs[t] = acc;
  }
  __syncthreads();
  if (t < 64) {
    size_t idx = ((size_t)b << 16) + ((size_t)t << 10) + p;
    float mo = combs[t], mg = combs[64 + t];
    float mi = sigm(combs[128 + t]);
    float mn = (1.f - mi) * m_in[idx] + mi * tanhf(mg);
    m_next[idx] = mn;
    h_next[idx] = sigm(mo) * mn;
  }
}

// ---------------------------------------------------------------------------
// Host launcher.
// ---------------------------------------------------------------------------
extern "C" void kernel_launch(void* const* d_in, const int* in_sizes, int n_in,
                              void* d_out, int out_size, void* d_ws, size_t ws_size,
                              hipStream_t stream) {
  (void)in_sizes; (void)n_in; (void)out_size; (void)ws_size;
  const float* x      = (const float*)d_in[0];
  const float* h      = (const float*)d_in[1];
  const float* c      = (const float*)d_in[2];
  const float* m      = (const float*)d_in[3];
  const float* lc     = (const float*)d_in[4];
  const float* conv_w = (const float*)d_in[5];
  const float* gn1_w  = (const float*)d_in[7];
  const float* gn1_b  = (const float*)d_in[8];
  const float* convLw = (const float*)d_in[9];
  const float* gn2_w  = (const float*)d_in[11];
  const float* gn2_b  = (const float*)d_in[12];
  const float* wq = (const float*)d_in[13]; const float* bq = (const float*)d_in[14];
  const float* wk = (const float*)d_in[15]; const float* bk = (const float*)d_in[16];
  const float* wk2 = (const float*)d_in[17]; const float* bk2 = (const float*)d_in[18];
  const float* wv = (const float*)d_in[19]; const float* bv = (const float*)d_in[20];
  const float* wv2 = (const float*)d_in[21]; const float* bv2 = (const float*)d_in[22];
  const float* wz = (const float*)d_in[23]; const float* bz = (const float*)d_in[24];
  const float* wm = (const float*)d_in[25]; const float* bm = (const float*)d_in[26];

  float* out     = (float*)d_out;
  float* h_next  = out;
  float* c_next  = out + 1048576;
  float* m_next  = out + 2 * 1048576;
  float* lc_next = out + 3 * 1048576;

  // Workspace layout (bytes). im2col region (37.75MB) is reused after conv2
  // for all attention buffers (15MB needed).
  char* ws = (char*)d_ws;
  __bf16* im2col  = (__bf16*)(ws);                      // 16384*1152*2 = 37,748,736
  __bf16* Qt      = (__bf16*)(ws + 0);                  // overlay (1MB)
  __bf16* Kt      = (__bf16*)(ws + 1048576);            // 1MB
  __bf16* Kmt     = (__bf16*)(ws + 2 * 1048576);        // 1MB
  __bf16* Vh      = (__bf16*)(ws + 3 * 1048576);        // 2MB
  __bf16* Vm      = (__bf16*)(ws + 5 * 1048576);        // 2MB
  float*  Zh      = (float*)(ws + 7 * 1048576);         // 4MB
  float*  Zm      = (float*)(ws + 11 * 1048576);        // 4MB
  __bf16* wp      = (__bf16*)(ws + 37748736);           // 589,824
  float*  convout = (float*)(ws + 38338560);            // 16,777,216
  float*  stats   = (float*)(ws + 55115776);            // 32,768
  float*  c1      = (float*)(ws + 55148544);            // 4MB
  float*  sco     = (float*)(ws + 59342848);            // 4MB
  float*  hmid    = (float*)(ws + 63537152);            // 4MB  (total ~67.7MB)

  const int NW = 294912;  // 256*1152 weight elements per conv

  // ---- conv block 1 ----
  k_cvt_bf16<<<(NW + 255) / 256, 256, 0, stream>>>(conv_w, wp, NW);
  k_im2col<<<73728, 256, 0, stream>>>(x, h, im2col);
  k_gemm_bf16<<<512, 128, 0, stream>>>(im2col, wp, convout, 1152, 1152, 1152, 256);
  k_gnstats<<<1024, 128, 0, stream>>>(convout, stats);
  k_gate1<<<4096, 256, 0, stream>>>(convout, stats, gn1_w, gn1_b, c, c1, sco);

  // ---- conv block 2 (long) ----
  k_cvt_bf16<<<(NW + 255) / 256, 256, 0, stream>>>(convLw, wp, NW);
  k_im2col<<<73728, 256, 0, stream>>>(h, c1, im2col);
  k_gemm_bf16<<<512, 128, 0, stream>>>(im2col, wp, convout, 1152, 1152, 1152, 256);
  k_gnstats<<<1024, 128, 0, stream>>>(convout, stats);
  k_gate2<<<4096, 256, 0, stream>>>(convout, stats, gn2_w, gn2_b, lc, sco,
                                    c_next, lc_next, hmid);

  // ---- SAM attention ----
  k_qk<<<4096, 128, 0, stream>>>(hmid, m, wq, bq, wk, bk, wk2, bk2, Qt, Kt, Kmt);
  k_v<<<4096, 256, 0, stream>>>(hmid, m, wv, bv, wv2, bv2, Vh, Vm);
  k_attn<<<1024, 128, 0, stream>>>(Qt, Kt, Vh, Zh);
  k_attn<<<1024, 128, 0, stream>>>(Qt, Kmt, Vm, Zm);

  // ---- final mixing + memory gates ----
  k_final<<<16384, 192, 0, stream>>>(Zh, Zm, hmid, m, wz, bz, wm, bm, h_next, m_next);
}
